// LocationSensitiveAttentionNaive_88759794139536
// MI455X (gfx1250) — matrix operand order repaired and checked by reading, not compiled, not run on hardware
//
#include <hip/hip_runtime.h>
#include <hip/hip_bf16.h>

// ---------------------------------------------------------------------------
// Location-sensitive attention, MI455X (gfx1250, wave32, WMMA).
//
// Shapes: B=64, T=2048, E=512, D=1024, H=128, C=32, K=31, PAD=15.
// Dominant cost: x @ V for both branches = 34 GFLOP, plus 2 full reads of the
// 256MB activation tensors (~1GB total -> ~43us HBM floor @ 23.3 TB/s).
// GEMM AI ~67 FLOP/B  => must run on v_wmma_f32_16x16x32_f16 to stay HBM-bound.
// Conv1d + (f @ U) is folded algebraically: g[b,t,h] = sum_k ap[b,t+k-15]*G[k,h]
// with G = F^T U  (31x128, precomputed, LDS-resident in the epilogue).
// ---------------------------------------------------------------------------

typedef __attribute__((ext_vector_type(16))) _Float16 v16h;
typedef __attribute__((ext_vector_type(8)))  float    v8f;

#define B_   64
#define T_   2048
#define E_   512
#define D_   1024
#define H_   128
#define C_   32
#define KW_  31
#define PAD_ 15

// ---------------------------------------------------------------------------
// Prep 1: pack V (E x H fp32) into f16 WMMA-B fragments.
// Layout: Vp[nt][kb][lane][e], e=0..15 halves (32B contiguous per lane).
//   lane<16 : K = kb*32 + {0..7, 16..23}[e] , N = nt*16 + lane
//   lane>=16: K = kb*32 + {8..15, 24..31}[e], N = nt*16 + (lane-16)
// ---------------------------------------------------------------------------
__global__ __launch_bounds__(256) void lsa_pack_v_kernel(
    const float* __restrict__ V, _Float16* __restrict__ Vp) {
  int idx = blockIdx.x * blockDim.x + threadIdx.x;   // 0 .. 8*16*32-1
  if (idx >= 8 * 16 * 32) return;
  int lane = idx & 31;
  int kb   = (idx >> 5) & 15;
  int nt   = idx >> 9;
  int n  = nt * 16 + (lane & 15);
  int k0 = kb * 32 + ((lane >> 4) ? 8 : 0);
  _Float16* o = Vp + ((size_t)idx << 4);
#pragma unroll
  for (int e = 0; e < 16; ++e) o[e] = (_Float16)V[(k0 + (e & 7) + ((e >> 3) << 4)) * H_ + n];
}

// ---------------------------------------------------------------------------
// Prep 2: base[b,h] = (dec @ W)[b,h] + Vb[h] for both branches. 64x128, tiny.
// ---------------------------------------------------------------------------
__global__ __launch_bounds__(128) void lsa_base_kernel(
    const float* __restrict__ dec,
    const float* __restrict__ W1, const float* __restrict__ Vb1,
    const float* __restrict__ W2, const float* __restrict__ Vb2,
    float* __restrict__ base1, float* __restrict__ base2) {
  int b = blockIdx.x, h = threadIdx.x;
  const float* dp = dec + (size_t)b * D_;
  float a1 = 0.f, a2 = 0.f;
  for (int d = 0; d < D_; ++d) {
    float dv = dp[d];
    a1 = fmaf(dv, W1[d * H_ + h], a1);
    a2 = fmaf(dv, W2[d * H_ + h], a2);
  }
  base1[b * H_ + h] = a1 + Vb1[h];
  base2[b * H_ + h] = a2 + Vb2[h];
}

// ---------------------------------------------------------------------------
// Prep 3: G[k,h] = sum_c F[c,0,k] * U[c,h]  (31x128, padded to 32 rows).
// ---------------------------------------------------------------------------
__global__ __launch_bounds__(128) void lsa_g_kernel(
    const float* __restrict__ F1, const float* __restrict__ U1,
    const float* __restrict__ F2, const float* __restrict__ U2,
    float* __restrict__ G1, float* __restrict__ G2) {
  int k = blockIdx.x;    // 0..31 (row 31 zero pad)
  int h = threadIdx.x;   // 0..127
  float g1 = 0.f, g2 = 0.f;
  if (k < KW_) {
    for (int c = 0; c < C_; ++c) {
      g1 = fmaf(F1[c * KW_ + k], U1[c * H_ + h], g1);
      g2 = fmaf(F2[c * KW_ + k], U2[c * H_ + h], g2);
    }
  }
  G1[k * H_ + h] = g1;
  G2[k * H_ + h] = g2;
}

// ---------------------------------------------------------------------------
// Main: scores for one branch.
//   s[b,t] = w . tanh( (x @ V)[b,t,:] + base[b,:] + g[b,t,:] ) + wb
// Grid: 1024 workgroups x 256 threads (8 waves). WG = 128 rows of one batch b.
// Each wave: 16x128 tile via 8 f32 WMMA accumulators, 16 k-steps of 32.
// ---------------------------------------------------------------------------
__global__ __launch_bounds__(256) void lsa_scores_wmma_kernel(
    const float*    __restrict__ x,      // [B*T, E]
    const _Float16* __restrict__ Vp,     // packed B fragments
    const float*    __restrict__ base,   // [B, H]
    const float*    __restrict__ G,      // [32, H]
    const float*    __restrict__ ap,     // [B, T] att_prev
    const float*    __restrict__ ww,     // [H]
    const float*    __restrict__ wb,     // scalar
    float*          __restrict__ s) {    // [B, T]
  __shared__ float Gs[32 * H_];
  __shared__ float aps[160];

  const int tid  = threadIdx.x;
  const int lane = tid & 31;
  const int wave = tid >> 5;
  const int blockRow = blockIdx.x;          // 0..1023
  const int b    = blockRow >> 4;           // T/128 = 16 row-blocks per batch
  const int tblk = (blockRow & 15) * 128;

  // Stage G (16KB) and the att_prev halo window (158 floats) into LDS.
  for (int i = tid; i < 32 * H_; i += 256) Gs[i] = G[i];
  for (int i = tid; i < 158; i += 256) {
    int t = tblk - PAD_ + i;
    aps[i] = (t >= 0 && t < T_) ? ap[(size_t)b * T_ + t] : 0.f;
  }
  __syncthreads();

  // A-fragment addressing (ISA 16-bit 16x32 A layout):
  // lane<16: row M=lane, K = {0..7, 16..23}; lane>=16: M=lane-16, K += 8.
  const int  mrow = wave * 16 + (lane & 15);
  const long grow = (long)blockRow * 128 + mrow;
  const float* xrow = x + grow * E_;
  const int  koff = (lane >> 4) ? 8 : 0;

  v8f acc[8];
#pragma unroll
  for (int nt = 0; nt < 8; ++nt)
#pragma unroll
    for (int e = 0; e < 8; ++e) acc[nt][e] = 0.f;

#pragma unroll 4
  for (int kb = 0; kb < 16; ++kb) {
    const float* xp = xrow + kb * 32 + koff;
    __builtin_prefetch(xp + 32, 0, 3);            // global_prefetch next slice
    float4 f0 = *(const float4*)(xp + 0);
    float4 f1 = *(const float4*)(xp + 4);
    float4 f2 = *(const float4*)(xp + 16);
    float4 f3 = *(const float4*)(xp + 20);
    v16h a;
    a[0]  = (_Float16)f0.x; a[1]  = (_Float16)f0.y;
    a[2]  = (_Float16)f0.z; a[3]  = (_Float16)f0.w;
    a[4]  = (_Float16)f1.x; a[5]  = (_Float16)f1.y;
    a[6]  = (_Float16)f1.z; a[7]  = (_Float16)f1.w;
    a[8]  = (_Float16)f2.x; a[9]  = (_Float16)f2.y;
    a[10] = (_Float16)f2.z; a[11] = (_Float16)f2.w;
    a[12] = (_Float16)f3.x; a[13] = (_Float16)f3.y;
    a[14] = (_Float16)f3.z; a[15] = (_Float16)f3.w;

#pragma unroll
    for (int nt = 0; nt < 8; ++nt) {
      v16h bf = *(const v16h*)(Vp + (((size_t)(nt * 16 + kb) * 32 + lane) << 4));
      acc[nt] = __builtin_amdgcn_wmma_f32_16x16x32_f16(
          false, a, false, bf, (short)0, acc[nt], false, false);
    }
  }

  // Epilogue: fused location term + bias + tanh + dot(w, .) row reduction.
  // C layout: acc[nt][r] -> M = wave*16 + r + (lane>=16 ? 8:0), N = nt*16+(lane&15)
  const float wbv = wb[0];
  const int nlow  = lane & 15;
  const int rhalf = (lane >> 4) << 3;
  float rowsum[8];
#pragma unroll
  for (int r = 0; r < 8; ++r) rowsum[r] = 0.f;

#pragma unroll
  for (int nt = 0; nt < 8; ++nt) {
    int n = nt * 16 + nlow;
    float basev = base[b * H_ + n];
    float wlv   = ww[n];
#pragma unroll
    for (int r = 0; r < 8; ++r) {
      int rib = wave * 16 + r + rhalf;          // row within 128-block
      float g = 0.f;
#pragma unroll
      for (int k = 0; k < KW_; ++k)
        g = fmaf(aps[rib + k], Gs[k * H_ + n], g);
      float v = tanhf(acc[nt][r] + basev + g);
      rowsum[r] = fmaf(v, wlv, rowsum[r]);
    }
  }

  // Reduce over N within each 16-lane half (rows live per half in wave32).
#pragma unroll
  for (int r = 0; r < 8; ++r) {
    float v = rowsum[r];
    v += __shfl_xor(v, 1, 32);
    v += __shfl_xor(v, 2, 32);
    v += __shfl_xor(v, 4, 32);
    v += __shfl_xor(v, 8, 32);
    if ((lane & 15) == 0) {
      int rib = wave * 16 + r + rhalf;
      s[(long)blockRow * 128 + rib] = v + wbv;
    }
  }
}

// ---------------------------------------------------------------------------
// Combine + softmax over T per batch row. 64 blocks x 256 threads, 8 t/thread.
// erg = sqrt(erg1*erg2), masked -> -inf. Writes attn to ws and to d_out tail.
// ---------------------------------------------------------------------------
__global__ __launch_bounds__(256) void lsa_softmax_kernel(
    const float* __restrict__ s1, const float* __restrict__ s2,
    const unsigned char* __restrict__ em, const unsigned char* __restrict__ pm,
    float* __restrict__ attn, float* __restrict__ out_attn) {
  __shared__ float red[256];
  const int b = blockIdx.x, tid = threadIdx.x;
  float vals[8];
  float mx = -__builtin_inff();
#pragma unroll
  for (int i = 0; i < 8; ++i) {
    long idx = (long)b * T_ + tid + i * 256;
    float v = (em[idx] || pm[idx]) ? -__builtin_inff() : sqrtf(s1[idx] * s2[idx]);
    vals[i] = v;
    mx = fmaxf(mx, v);
  }
  red[tid] = mx; __syncthreads();
  for (int st = 128; st > 0; st >>= 1) {
    if (tid < st) red[tid] = fmaxf(red[tid], red[tid + st]);
    __syncthreads();
  }
  mx = red[0]; __syncthreads();
  float sum = 0.f;
#pragma unroll
  for (int i = 0; i < 8; ++i) { vals[i] = __expf(vals[i] - mx); sum += vals[i]; }
  red[tid] = sum; __syncthreads();
  for (int st = 128; st > 0; st >>= 1) {
    if (tid < st) red[tid] += red[tid + st];
    __syncthreads();
  }
  float inv = 1.f / red[0];
#pragma unroll
  for (int i = 0; i < 8; ++i) {
    long idx = (long)b * T_ + tid + i * 256;
    float a = vals[i] * inv;
    attn[idx] = a;
    out_attn[idx] = a;
  }
}

// ---------------------------------------------------------------------------
// Context pass 1: per (b, t-chunk of 256) partial sums over both tensors.
// Fully coalesced streaming read of the 2x256MB tensors (second HBM pass).
// part layout: [B][8][2][E]
// ---------------------------------------------------------------------------
__global__ __launch_bounds__(256) void lsa_ctx_partial_kernel(
    const float* __restrict__ eo, const float* __restrict__ po,
    const float* __restrict__ attn, float* __restrict__ part) {
  const int b = blockIdx.x, tc = blockIdx.y, tid = threadIdx.x;
  const int e0 = tid, e1 = tid + 256;
  float a0 = 0.f, a1 = 0.f, p0 = 0.f, p1 = 0.f;
  const long baseX = ((long)b * T_ + (long)tc * 256) * E_;
  const float* apw = attn + (size_t)b * T_ + tc * 256;
  for (int t = 0; t < 256; ++t) {
    float w = apw[t];
    const float* er = eo + baseX + (long)t * E_;
    const float* pr = po + baseX + (long)t * E_;
    a0 = fmaf(er[e0], w, a0); a1 = fmaf(er[e1], w, a1);
    p0 = fmaf(pr[e0], w, p0); p1 = fmaf(pr[e1], w, p1);
  }
  float* o = part + ((long)(b * 8 + tc) * 2) * E_;
  o[e0]      = a0; o[e1]      = a1;
  o[E_ + e0] = p0; o[E_ + e1] = p1;
}

// Context pass 2: reduce 8 t-chunk partials -> d_out[0:2*B*E).
__global__ __launch_bounds__(256) void lsa_ctx_reduce_kernel(
    const float* __restrict__ part, float* __restrict__ out) {
  int idx = blockIdx.x * blockDim.x + threadIdx.x;    // 0 .. B*E-1
  if (idx >= B_ * E_) return;
  int b = idx / E_, e = idx % E_;
  float se = 0.f, sp = 0.f;
  for (int tc = 0; tc < 8; ++tc) {
    se += part[((long)(b * 8 + tc) * 2 + 0) * E_ + e];
    sp += part[((long)(b * 8 + tc) * 2 + 1) * E_ + e];
  }
  out[idx] = se;
  out[B_ * E_ + idx] = sp;
}

// ---------------------------------------------------------------------------
extern "C" void kernel_launch(void* const* d_in, const int* in_sizes, int n_in,
                              void* d_out, int out_size, void* d_ws, size_t ws_size,
                              hipStream_t stream) {
  const float* energy_outs = (const float*)d_in[0];
  const float* pitch_outs  = (const float*)d_in[2];
  const float* dec         = (const float*)d_in[4];
  const float* att_prev    = (const float*)d_in[5];
  const unsigned char* em  = (const unsigned char*)d_in[6];
  const unsigned char* pm  = (const unsigned char*)d_in[7];
  const float* V1w = (const float*)d_in[8];
  const float* V1b = (const float*)d_in[9];
  const float* W1w = (const float*)d_in[10];
  const float* U1w = (const float*)d_in[11];
  const float* F1w = (const float*)d_in[12];
  const float* w1w = (const float*)d_in[13];
  const float* w1b = (const float*)d_in[14];
  const float* V2w = (const float*)d_in[15];
  const float* V2b = (const float*)d_in[16];
  const float* W2w = (const float*)d_in[17];
  const float* U2w = (const float*)d_in[18];
  const float* F2w = (const float*)d_in[19];
  const float* w2w = (const float*)d_in[20];
  const float* w2b = (const float*)d_in[21];
  float* out = (float*)d_out;

  // Workspace carve-up (~3.9 MB total).
  char* w = (char*)d_ws;
  _Float16* V1p = (_Float16*)w;                   // 65536 halves
  _Float16* V2p = V1p + 65536;                    // 65536 halves
  float* base1 = (float*)(w + 262144);            // [64,128]
  float* base2 = base1 + 8192;
  float* G1    = base2 + 8192;                    // [32,128]
  float* G2    = G1 + 4096;
  float* s1    = G2 + 4096;                       // [64,2048]
  float* s2    = s1 + 131072;
  float* attn  = s2 + 131072;                     // [64,2048]
  float* part  = attn + 131072;                   // [64][8][2][512]

  lsa_pack_v_kernel<<<16, 256, 0, stream>>>(V1w, V1p);
  lsa_pack_v_kernel<<<16, 256, 0, stream>>>(V2w, V2p);
  lsa_base_kernel<<<64, 128, 0, stream>>>(dec, W1w, V1b, W2w, V2b, base1, base2);
  lsa_g_kernel<<<32, 128, 0, stream>>>(F1w, U1w, F2w, U2w, G1, G2);

  lsa_scores_wmma_kernel<<<1024, 256, 0, stream>>>(
      energy_outs, V1p, base1, G1, att_prev, w1w, w1b, s1);
  lsa_scores_wmma_kernel<<<1024, 256, 0, stream>>>(
      pitch_outs, V2p, base2, G2, att_prev, w2w, w2b, s2);

  lsa_softmax_kernel<<<64, 256, 0, stream>>>(
      s1, s2, em, pm, attn, out + (size_t)B_ * E_ * 2);

  lsa_ctx_partial_kernel<<<dim3(64, 8), 256, 0, stream>>>(
      energy_outs, pitch_outs, attn, part);
  lsa_ctx_reduce_kernel<<<128, 256, 0, stream>>>(part, out);
}
